// GRUMemoryUpdater_1511828488873
// MI455X (gfx1250) — matrix operand order repaired and checked by reading, not compiled, not run on hardware
//
#include <hip/hip_runtime.h>
#include <hip/hip_bf16.h>

// ---- problem constants (from reference) ------------------------------------
#define D_MEM   172
#define KPAD    192          // 172 padded to 6*32 for K=32 bf16 WMMA steps
#define N3      516          // 3 * D_MEM gate rows
#define NPAD    528          // 516 padded to 33*16
#define N_UPD   200000       // exactly 12500 tiles of 16
#define N_NODES 1000000

typedef __attribute__((ext_vector_type(16))) __bf16 v16bf;
typedef __attribute__((ext_vector_type(8)))  float  v8f;
typedef __attribute__((ext_vector_type(4)))  float  v4f;   // clang vector: OK for NT builtins

union ABfrag {
  v16bf v;
  uint4 u4[2];
};

// ---- 1) weights f32 -> zero-padded bf16 [NPAD][KPAD] -----------------------
__global__ void convert_weights(const float* __restrict__ wih,
                                const float* __restrict__ whh,
                                __bf16* __restrict__ wihb,
                                __bf16* __restrict__ whhb) {
  int idx = blockIdx.x * blockDim.x + threadIdx.x;
  const int per = NPAD * KPAD;
  if (idx >= 2 * per) return;
  const float* src = (idx < per) ? wih : whh;
  __bf16*      dst = (idx < per) ? wihb : whhb;
  int i = (idx < per) ? idx : idx - per;
  int r = i / KPAD, k = i % KPAD;
  float v = (r < N3 && k < D_MEM) ? src[r * D_MEM + k] : 0.0f;
  dst[i] = (__bf16)v;
}

// ---- 2) bulk copy memory + last_update into d_out (bandwidth bound) --------
// 688 MB in + 688 MB out streams >> 192 MB L2: use non-temporal loads/stores
// (TH=NT cache policy) so the stream doesn't evict the L2-resident weights
// and the gather/scatter lines the GRU kernel re-touches.
__global__ void copy_bulk(const v4f* __restrict__ mem,
                          const v4f* __restrict__ lu,
                          v4f* __restrict__ outMem,
                          v4f* __restrict__ outLu) {
  const long long NM = (long long)N_NODES * D_MEM / 4;  // 43,000,000
  const long long NL = N_NODES / 4;                     // 250,000
  long long stride = (long long)gridDim.x * blockDim.x;
  for (long long i = (long long)blockIdx.x * blockDim.x + threadIdx.x;
       i < NM + NL; i += stride) {
    if (i < NM) {
      v4f v = __builtin_nontemporal_load(&mem[i]);
      __builtin_nontemporal_store(v, &outMem[i]);
    } else {
      v4f v = __builtin_nontemporal_load(&lu[i - NM]);
      __builtin_nontemporal_store(v, &outLu[i - NM]);
    }
  }
}

// ---- 3) fused gather + GRUCell (bf16 WMMA) + scatter -----------------------
// Block = 352 threads = 11 wave32; wave w owns hidden columns [16w, 16w+16).
__global__ __launch_bounds__(352) void gru_update(
    const float*     __restrict__ memory,
    const long long* __restrict__ ids,
    const float*     __restrict__ msgs,
    const float*     __restrict__ ts,
    const __bf16*    __restrict__ wihb,   // [NPAD][KPAD] bf16
    const __bf16*    __restrict__ whhb,   // [NPAD][KPAD] bf16
    const float*     __restrict__ bih,
    const float*     __restrict__ bhh,
    float*           __restrict__ outMem,
    float*           __restrict__ outLu)
{
  __shared__ __align__(16) __bf16   Xb[16 * KPAD];   // messages, bf16, K-padded
  __shared__ __align__(16) __bf16   Hb[16 * KPAD];   // gathered h, bf16, K-padded
  __shared__ __align__(16) float    Hf[16 * 176];    // gathered h, f32 (stride 176)
  __shared__               long long rowId[16];

  const int tid     = threadIdx.x;
  const int tile    = blockIdx.x;
  const int rowBase = tile * 16;

  if (tid < 16) {
    long long id = ids[rowBase + tid];
    rowId[tid] = id;
    outLu[id] = ts[rowBase + tid];            // scatter timestamps
  }
  __syncthreads();

  // stage X and H tiles (f32 -> bf16, zero pad K to 192).
  // Messages are touched exactly once per launch -> non-temporal load.
  for (int i = tid; i < 16 * KPAD; i += 352) {
    int m = i / KPAD, k = i % KPAD;
    float xv = (k < D_MEM)
                 ? __builtin_nontemporal_load(&msgs[(long long)(rowBase + m) * D_MEM + k])
                 : 0.0f;
    float hv = (k < D_MEM) ? memory[rowId[m] * D_MEM + k] : 0.0f;
    Xb[i] = (__bf16)xv;
    Hb[i] = (__bf16)hv;
    if (k < 176) Hf[m * 176 + k] = (k < D_MEM) ? hv : 0.0f;
  }
  __syncthreads();

  const int wave = tid >> 5;        // 0..10 -> column tile
  const int lane = tid & 31;
  const int lh   = lane >> 4;       // lane half (ISA 16-bit fragment layout)
  const int l15  = lane & 15;
  const int nCol = wave * 16 + l15; // hidden column (0..175)

  v8f aIR = {}, aIZ = {}, aIN = {};   // gi accumulators (r,z,n gate rows)
  v8f aHR = {}, aHZ = {}, aHN = {};   // gh accumulators

  const unsigned int* XbU = (const unsigned int*)Xb;
  const unsigned int* HbU = (const unsigned int*)Hb;
  const unsigned int* WIu = (const unsigned int*)wihb;
  const unsigned int* WHu = (const unsigned int*)whhb;

  // W rows (= output gate columns) for this lane
  const unsigned int* wiR = WIu + (long long)(0 * D_MEM + nCol) * (KPAD / 2);
  const unsigned int* wiZ = WIu + (long long)(1 * D_MEM + nCol) * (KPAD / 2);
  const unsigned int* wiN = WIu + (long long)(2 * D_MEM + nCol) * (KPAD / 2);
  const unsigned int* whR = WHu + (long long)(0 * D_MEM + nCol) * (KPAD / 2);
  const unsigned int* whZ = WHu + (long long)(1 * D_MEM + nCol) * (KPAD / 2);
  const unsigned int* whN = WHu + (long long)(2 * D_MEM + nCol) * (KPAD / 2);

#pragma unroll
  for (int ks = 0; ks < 6; ++ks) {
    const int k2 = ks * 16;                 // K-step base in 32-bit units
    // A fragments: lanes 0-15 hold K 0-15, lanes 16-31 hold K 16-31 (contiguous)
    ABfrag ax, ah;
    ax.u4[0] = *(const uint4*)(XbU + l15 * 96 + k2 + 8 * lh);
    ax.u4[1] = *(const uint4*)(XbU + l15 * 96 + k2 + 4 + 8 * lh);
    ah.u4[0] = *(const uint4*)(HbU + l15 * 96 + k2 + 8 * lh);
    ah.u4[1] = *(const uint4*)(HbU + l15 * 96 + k2 + 4 + 8 * lh);

    // B fragments: lanes 0-15 hold K 0-15, lanes 16-31 hold K 16-31 (contiguous)
    const int bo = k2 + 8 * lh;
    ABfrag bir, biz, bin, bhr, bhz, bhn;
    bir.u4[0] = *(const uint4*)(wiR + bo); bir.u4[1] = *(const uint4*)(wiR + bo + 4);
    biz.u4[0] = *(const uint4*)(wiZ + bo); biz.u4[1] = *(const uint4*)(wiZ + bo + 4);
    bin.u4[0] = *(const uint4*)(wiN + bo); bin.u4[1] = *(const uint4*)(wiN + bo + 4);
    bhr.u4[0] = *(const uint4*)(whR + bo); bhr.u4[1] = *(const uint4*)(whR + bo + 4);
    bhz.u4[0] = *(const uint4*)(whZ + bo); bhz.u4[1] = *(const uint4*)(whZ + bo + 4);
    bhn.u4[0] = *(const uint4*)(whN + bo); bhn.u4[1] = *(const uint4*)(whN + bo + 4);

    aIR = __builtin_amdgcn_wmma_f32_16x16x32_bf16(false, ax.v, false, bir.v, (short)0, aIR, false, false);
    aIZ = __builtin_amdgcn_wmma_f32_16x16x32_bf16(false, ax.v, false, biz.v, (short)0, aIZ, false, false);
    aIN = __builtin_amdgcn_wmma_f32_16x16x32_bf16(false, ax.v, false, bin.v, (short)0, aIN, false, false);
    aHR = __builtin_amdgcn_wmma_f32_16x16x32_bf16(false, ah.v, false, bhr.v, (short)0, aHR, false, false);
    aHZ = __builtin_amdgcn_wmma_f32_16x16x32_bf16(false, ah.v, false, bhz.v, (short)0, aHZ, false, false);
    aHN = __builtin_amdgcn_wmma_f32_16x16x32_bf16(false, ah.v, false, bhn.v, (short)0, aHN, false, false);
  }

  // epilogue: gates + scatter. C layout: lane l15 -> N, VGPR j -> M = j + 8*lh.
  const bool colOK = (nCol < D_MEM);
  float cIR = 0.f, cIZ = 0.f, cIN = 0.f, cHR = 0.f, cHZ = 0.f, cHN = 0.f;
  if (colOK) {
    cIR = bih[nCol];             cHR = bhh[nCol];
    cIZ = bih[D_MEM + nCol];     cHZ = bhh[D_MEM + nCol];
    cIN = bih[2 * D_MEM + nCol]; cHN = bhh[2 * D_MEM + nCol];
  }
#pragma unroll
  for (int j = 0; j < 8; ++j) {
    const int m = j + 8 * lh;
    float r  = 1.0f / (1.0f + __expf(-(aIR[j] + cIR + aHR[j] + cHR)));
    float z  = 1.0f / (1.0f + __expf(-(aIZ[j] + cIZ + aHZ[j] + cHZ)));
    float nn = tanhf(aIN[j] + cIN + r * (aHN[j] + cHN));
    float h  = Hf[m * 176 + nCol];
    float hnew = nn + z * (h - nn);
    if (colOK) outMem[rowId[m] * D_MEM + nCol] = hnew;   // coalesced 64B runs
  }
}

// ---- launcher --------------------------------------------------------------
extern "C" void kernel_launch(void* const* d_in, const int* in_sizes, int n_in,
                              void* d_out, int out_size, void* d_ws, size_t ws_size,
                              hipStream_t stream) {
  const float*     memory = (const float*)d_in[0];
  const float*     lastup = (const float*)d_in[1];
  const long long* ids    = (const long long*)d_in[2];
  const float*     msgs   = (const float*)d_in[3];
  const float*     ts     = (const float*)d_in[4];
  const float*     wih    = (const float*)d_in[5];
  const float*     whh    = (const float*)d_in[6];
  const float*     bih    = (const float*)d_in[7];
  const float*     bhh    = (const float*)d_in[8];

  float* outMem = (float*)d_out;                            // [N_NODES,1,D]
  float* outLu  = outMem + (long long)N_NODES * D_MEM;      // [N_NODES]

  // workspace: two padded bf16 weight matrices (2 * 528*192*2 B = 406 KB)
  __bf16* wihb = (__bf16*)d_ws;
  __bf16* whhb = wihb + NPAD * KPAD;

  // 1) convert weights (tiny)
  const int convN = 2 * NPAD * KPAD;
  convert_weights<<<(convN + 255) / 256, 256, 0, stream>>>(wih, whh, wihb, whhb);

  // 2) bulk copy of memory + last_update into output (non-temporal stream)
  copy_bulk<<<8192, 256, 0, stream>>>((const v4f*)memory, (const v4f*)lastup,
                                      (v4f*)outMem, (v4f*)outLu);

  // 3) fused gather + GRU + scatter; 12500 M-tiles, 11 waves each
  gru_update<<<N_UPD / 16, 352, 0, stream>>>(memory, ids, msgs, ts,
                                             wihb, whhb, bih, bhh,
                                             outMem, outLu);
}
